// DiffusionInteractionBlock_2370821947765
// MI455X (gfx1250) — matrix operand order repaired and checked by reading, not compiled, not run on hardware
//
#include <hip/hip_runtime.h>
#include <hip/hip_fp16.h>
#include <cmath>

typedef __attribute__((ext_vector_type(16))) _Float16 v16h;
typedef __attribute__((ext_vector_type(8)))  _Float16 v8h;
typedef __attribute__((ext_vector_type(8)))  float    v8f;

#define NNODE 10000
#define NEDGE 200000

__device__ __forceinline__ float silu_f(float x) {
  return x / (1.0f + __expf(-x));
}

// Build A fragment (16x32 f16, ISA layout) from contiguous f16 row segment.
// p must already point at Arow + k0 + koff, 16B aligned.
__device__ __forceinline__ v16h load_a_h(const _Float16* p) {
  v8h lo = *(const v8h*)(p);
  v8h hi = *(const v8h*)(p + 16);
  v16h a;
#pragma unroll
  for (int i = 0; i < 8; ++i) { a[i] = lo[i]; a[8 + i] = hi[i]; }
  return a;
}

// Build A fragment from f32 source with element stride (converted to f16).
// p points at element (k0+koff) of the logical row.
__device__ __forceinline__ v16h load_a_f32(const float* p, int stride) {
  v16h a;
#pragma unroll
  for (int i = 0; i < 8; ++i) {
    a[i]     = (_Float16)p[i * stride];
    a[8 + i] = (_Float16)p[(16 + i) * stride];
  }
  return a;
}

// acc[8] += A(16xK) @ B(Kx128).  Bt is (128 x K) row-major f16 (B transposed).
// aload(k0) returns the A fragment for K-step k0.
template <int K, class ALoad>
__device__ __forceinline__ void wmma_tile(v8f acc[8], ALoad aload,
                                          const _Float16* __restrict__ Bt,
                                          int colb, int koff2) {
#pragma unroll
  for (int ks = 0; ks < K / 32; ++ks) {
    const int k0 = ks * 32;
    v16h a = aload(k0);
#pragma unroll
    for (int nt = 0; nt < 8; ++nt) {
      v16h b = load_a_h(Bt + (size_t)(nt * 16 + colb) * K + k0 + koff2 / 2);
      acc[nt] = __builtin_amdgcn_wmma_f32_16x16x32_f16(
          false, a, false, b, (short)0, acc[nt], false, false);
    }
  }
}

// ---------------- weight prep: transpose (KxNn) f32 -> (NnxK) f16, scaled ----
__global__ void k_transpose_f16(const float* __restrict__ src,
                                _Float16* __restrict__ dst,
                                int K, int Nn, float scale) {
  int idx = blockIdx.x * blockDim.x + threadIdx.x;
  if (idx >= K * Nn) return;
  int k = idx / Nn, n = idx - k * Nn;
  dst[(size_t)n * K + k] = (_Float16)(src[idx] * scale);
}

__global__ void k_zero(float* __restrict__ p, int n) {
  int i = blockIdx.x * blockDim.x + threadIdx.x;
  int st = gridDim.x * blockDim.x;
  for (; i < n; i += st) p[i] = 0.0f;
}

// ---------------- node transforms: ns(f16), s_up(f32), v_up(3 planes f32) ---
__global__ void __launch_bounds__(256) k_node_up(
    const float* __restrict__ nf,
    const _Float16* __restrict__ Wst, const _Float16* __restrict__ WupSt,
    const _Float16* __restrict__ WupVt,
    _Float16* __restrict__ ns_f16, float* __restrict__ s_up,
    float* __restrict__ v_up) {
  const int lane = threadIdx.x & 31;
  const int wave = threadIdx.x >> 5;
  const int base = (blockIdx.x * 8 + wave) * 16;
  if (base >= NNODE) return;
  const int colb  = lane & 15;
  const int koff  = (lane >> 4) * 8;
  const int koff2 = (lane >> 4) * 16;
  const int mb    = (lane >> 4) * 8;
  int rowa = base + colb;
  if (rowa > NNODE - 1) rowa = NNODE - 1;
  const float* nfrow = nf + (size_t)rowa * 512;

  // ns = (s @ Ws) * 1/sqrt(C)  (scale baked into Wst) -> f16
  {
    v8f acc[8] = {};
    auto loadS = [&](int k0) { return load_a_f32(nfrow + k0 + koff, 1); };
    wmma_tile<128>(acc, loadS, Wst, colb, koff2);
#pragma unroll
    for (int r = 0; r < 8; ++r) {
      int row2 = base + mb + r;
      if (row2 < NNODE) {
#pragma unroll
        for (int nt = 0; nt < 8; ++nt)
          ns_f16[(size_t)row2 * 128 + nt * 16 + colb] = (_Float16)acc[nt][r];
      }
    }
  }
  // s_up = (s @ Wup_s) * 1/sqrt(C) -> f32
  {
    v8f acc[8] = {};
    auto loadS = [&](int k0) { return load_a_f32(nfrow + k0 + koff, 1); };
    wmma_tile<128>(acc, loadS, WupSt, colb, koff2);
#pragma unroll
    for (int r = 0; r < 8; ++r) {
      int row2 = base + mb + r;
      if (row2 < NNODE) {
#pragma unroll
        for (int nt = 0; nt < 8; ++nt)
          s_up[(size_t)row2 * 128 + nt * 16 + colb] = acc[nt][r];
      }
    }
  }
  // v_up_c = (v[:,:,c] @ Wup_v) * 1/sqrt(C) -> f32, plane per component
#pragma unroll
  for (int c = 0; c < 3; ++c) {
    v8f acc[8] = {};
    auto loadV = [&](int k0) {
      return load_a_f32(nfrow + 128 + 3 * (k0 + koff) + c, 3);
    };
    wmma_tile<128>(acc, loadV, WupVt, colb, koff2);
    float* dst = v_up + (size_t)c * NNODE * 128;
#pragma unroll
    for (int r = 0; r < 8; ++r) {
      int row2 = base + mb + r;
      if (row2 < NNODE) {
#pragma unroll
        for (int nt = 0; nt < 8; ++nt)
          dst[(size_t)row2 * 128 + nt * 16 + colb] = acc[nt][r];
      }
    }
  }
}

// ---------------- edge MLP + pk-f16 message scatter --------------------------
__global__ void __launch_bounds__(256) k_edge(
    const _Float16* __restrict__ ns_f16,
    const float* __restrict__ s_up, const float* __restrict__ v_up,
    const float* __restrict__ edge_attrs, const float* __restrict__ edge_feats,
    const float* __restrict__ lengths,
    const int* __restrict__ sender, const int* __restrict__ receiver,
    const float* __restrict__ W1f, const float* __restrict__ b1,
    const float* __restrict__ b2,
    const _Float16* __restrict__ W1t, const _Float16* __restrict__ W2t,
    const _Float16* __restrict__ W3t,
    _Float16* __restrict__ msg_s, _Float16* __restrict__ msg_v) {
  __shared__ _Float16 smem[8 * 16 * 128];  // 4KB per wave, reused h1 -> h2
  const int lane  = threadIdx.x & 31;
  const int wave  = threadIdx.x >> 5;
  const int ebase = (blockIdx.x * 8 + wave) * 16;
  if (ebase >= NEDGE) return;
  const int colb  = lane & 15;
  const int koff  = (lane >> 4) * 8;
  const int koff2 = (lane >> 4) * 16;
  const int mb    = (lane >> 4) * 8;
  _Float16* hbuf = smem + wave * 2048;

  int rowe = ebase + colb;
  if (rowe > NEDGE - 1) rowe = NEDGE - 1;
  const int snd_a = sender[rowe];
  const int rcv_a = receiver[rowe];

  // ---- layer 1: [ns[snd] | ns[rcv]] (16x256) @ W1 -> 16x128
  v8f acc1[8] = {};
  auto loadA1 = [&](int k0) {
    const _Float16* src =
        ns_f16 + (size_t)(k0 < 128 ? snd_a : rcv_a) * 128 + (k0 & 127) + koff;
    return load_a_h(src);
  };
  wmma_tile<256>(acc1, loadA1, W1t, colb, koff2);

  float w1a[8], w1b[8], bb[8];
#pragma unroll
  for (int nt = 0; nt < 8; ++nt) {
    const int ch = nt * 16 + colb;
    w1a[nt] = W1f[256 * 128 + ch];  // edge_feats coefficient row
    w1b[nt] = W1f[257 * 128 + ch];  // lengths coefficient row
    bb[nt]  = b1[ch];
  }
#pragma unroll
  for (int r = 0; r < 8; ++r) {
    int eg = ebase + mb + r;
    if (eg > NEDGE - 1) eg = NEDGE - 1;
    const float ef = edge_feats[eg];
    const float ln = lengths[eg];
#pragma unroll
    for (int nt = 0; nt < 8; ++nt) {
      float x = acc1[nt][r] + ef * w1a[nt] + ln * w1b[nt] + bb[nt];
      hbuf[(mb + r) * 128 + nt * 16 + colb] = (_Float16)silu_f(x);
    }
  }

  // ---- layer 2: h (16x128) @ W2 -> 16x128 (LDS reused, wave-local in-order DS)
  v8f acc2[8] = {};
  const _Float16* hrow = hbuf + colb * 128;
  auto loadA2 = [&](int k0) { return load_a_h(hrow + k0 + koff); };
  wmma_tile<128>(acc2, loadA2, W2t, colb, koff2);
#pragma unroll
  for (int nt = 0; nt < 8; ++nt) bb[nt] = b2[nt * 16 + colb];
#pragma unroll
  for (int r = 0; r < 8; ++r) {
#pragma unroll
    for (int nt = 0; nt < 8; ++nt) {
      float x = acc2[nt][r] + bb[nt];
      hbuf[(mb + r) * 128 + nt * 16 + colb] = (_Float16)silu_f(x);
    }
  }

  // ---- layer 3 in 4 chunks (w00,w11,w01,w10) + pk-f16 message scatter
  _Float16* msgv0 = msg_v;
  _Float16* msgv1 = msg_v + (size_t)NNODE * 256;
  _Float16* msgv2 = msg_v + (size_t)2 * NNODE * 256;
  const float* vup0 = v_up;
  const float* vup1 = v_up + (size_t)NNODE * 128;
  const float* vup2 = v_up + (size_t)2 * NNODE * 128;
  auto loadA3 = [&](int k0) { return load_a_h(hrow + k0 + koff); };
  const bool evenlane = (lane & 1) == 0;

#pragma unroll
  for (int q = 0; q < 4; ++q) {
    v8f acc3[8] = {};
    wmma_tile<128>(acc3, loadA3, W3t + (size_t)q * 128 * 128, colb, koff2);
#pragma unroll
    for (int r = 0; r < 8; ++r) {
      const int eg = ebase + mb + r;
      if (eg >= NEDGE) continue;  // pairs (lane, lane^1) share eg -> safe
      const int sn = sender[eg];
      const int rc = receiver[eg];
      const float a0 = edge_attrs[(size_t)eg * 4 + 0];
      const float a1 = edge_attrs[(size_t)eg * 4 + 1];
      const float a2 = edge_attrs[(size_t)eg * 4 + 2];
      const float a3 = edge_attrs[(size_t)eg * 4 + 3];
      const size_t sb = (size_t)sn * 128;
      const size_t rb = (size_t)rc * 256;
#pragma unroll
      for (int nt = 0; nt < 8; ++nt) {
        const int ch  = nt * 16 + colb;  // even lane -> even ch
        const float w = acc3[nt][r];
        if (q == 0) {                          // m_0e_a
          float m  = w * s_up[sb + ch] * a0;
          float mo = __shfl_xor(m, 1);
          if (evenlane)
            unsafeAtomicAdd((__half2*)&msg_s[rb + ch], __floats2half2_rn(m, mo));
        } else if (q == 1) {                   // m_0e_b
          float dot = vup0[sb + ch] * a1 + vup1[sb + ch] * a2 + vup2[sb + ch] * a3;
          float m   = w * dot * 0.5773502691896258f;
          float mo  = __shfl_xor(m, 1);
          if (evenlane)
            unsafeAtomicAdd((__half2*)&msg_s[rb + 128 + ch], __floats2half2_rn(m, mo));
        } else if (q == 2) {                   // m_1o_a
          float t  = w * s_up[sb + ch];
          float to = __shfl_xor(t, 1);
          if (evenlane) {
            unsafeAtomicAdd((__half2*)&msgv0[rb + ch], __floats2half2_rn(t * a1, to * a1));
            unsafeAtomicAdd((__half2*)&msgv1[rb + ch], __floats2half2_rn(t * a2, to * a2));
            unsafeAtomicAdd((__half2*)&msgv2[rb + ch], __floats2half2_rn(t * a3, to * a3));
          }
        } else {                               // m_1o_b
          float t  = w * a0;
          float to = __shfl_xor(t, 1);
          if (evenlane) {
            float2 v0 = *(const float2*)&vup0[sb + ch];
            float2 v1 = *(const float2*)&vup1[sb + ch];
            float2 v2 = *(const float2*)&vup2[sb + ch];
            unsafeAtomicAdd((__half2*)&msgv0[rb + 128 + ch], __floats2half2_rn(t * v0.x, to * v0.y));
            unsafeAtomicAdd((__half2*)&msgv1[rb + 128 + ch], __floats2half2_rn(t * v1.x, to * v1.y));
            unsafeAtomicAdd((__half2*)&msgv2[rb + 128 + ch], __floats2half2_rn(t * v2.x, to * v2.y));
          }
        }
      }
    }
  }
}

// ---------------- readout: msg(f16) @ L  -> out (N,128,4) --------------------
__global__ void __launch_bounds__(256) k_out(
    const _Float16* __restrict__ msg_s, const _Float16* __restrict__ msg_v,
    const _Float16* __restrict__ L_st, const _Float16* __restrict__ L_vt,
    float* __restrict__ out) {
  const int lane = threadIdx.x & 31;
  const int wave = threadIdx.x >> 5;
  const int base = (blockIdx.x * 8 + wave) * 16;
  if (base >= NNODE) return;
  const int colb  = lane & 15;
  const int koff  = (lane >> 4) * 8;
  const int koff2 = (lane >> 4) * 16;
  const int mb    = (lane >> 4) * 8;
  int rowa = base + colb;
  if (rowa > NNODE - 1) rowa = NNODE - 1;

#pragma unroll
  for (int comp = 0; comp < 4; ++comp) {
    const _Float16* src =
        (comp == 0) ? msg_s : (msg_v + (size_t)(comp - 1) * NNODE * 256);
    const _Float16* Bt = (comp == 0) ? L_st : L_vt;
    const _Float16* arow = src + (size_t)rowa * 256;
    v8f acc[8] = {};
    auto loadA = [&](int k0) { return load_a_h(arow + k0 + koff); };
    wmma_tile<256>(acc, loadA, Bt, colb, koff2);
#pragma unroll
    for (int r = 0; r < 8; ++r) {
      int row2 = base + mb + r;
      if (row2 < NNODE) {
#pragma unroll
        for (int nt = 0; nt < 8; ++nt)
          out[((size_t)row2 * 128 + nt * 16 + colb) * 4 + comp] = acc[nt][r];
      }
    }
  }
}

extern "C" void kernel_launch(void* const* d_in, const int* in_sizes, int n_in,
                              void* d_out, int out_size, void* d_ws, size_t ws_size,
                              hipStream_t stream) {
  (void)in_sizes; (void)n_in; (void)out_size; (void)ws_size;
  const float* node_feats = (const float*)d_in[0];
  const float* edge_attrs = (const float*)d_in[1];
  const float* edge_feats = (const float*)d_in[2];
  const float* lengths    = (const float*)d_in[3];
  const int*   edge_index = (const int*)d_in[4];
  const float* Ws    = (const float*)d_in[5];
  const float* Wup_s = (const float*)d_in[6];
  const float* Wup_v = (const float*)d_in[7];
  const float* W1    = (const float*)d_in[8];
  const float* b1    = (const float*)d_in[9];
  const float* W2    = (const float*)d_in[10];
  const float* b2    = (const float*)d_in[11];
  const float* W3    = (const float*)d_in[12];
  const float* L_s   = (const float*)d_in[13];
  const float* L_v   = (const float*)d_in[14];
  float* out = (float*)d_out;

  char* ws = (char*)d_ws;
  size_t off = 0;
  auto alloc = [&](size_t bytes) -> void* {
    void* p = ws + off;
    off = (off + bytes + 255) & ~(size_t)255;
    return p;
  };
  _Float16* ns_f16 = (_Float16*)alloc((size_t)NNODE * 128 * 2);
  float*    s_up   = (float*)alloc((size_t)NNODE * 128 * 4);
  float*    v_up   = (float*)alloc((size_t)3 * NNODE * 128 * 4);
  _Float16* msg    = (_Float16*)alloc((size_t)4 * NNODE * 256 * 2); // msg_s|msgv0|msgv1|msgv2 (f16)
  _Float16* Wst    = (_Float16*)alloc(128 * 128 * 2);
  _Float16* WupSt  = (_Float16*)alloc(128 * 128 * 2);
  _Float16* WupVt  = (_Float16*)alloc(128 * 128 * 2);
  _Float16* W1t    = (_Float16*)alloc(128 * 256 * 2);
  _Float16* W2t    = (_Float16*)alloc(128 * 128 * 2);
  _Float16* W3t    = (_Float16*)alloc(512 * 128 * 2);
  _Float16* L_st   = (_Float16*)alloc(128 * 256 * 2);
  _Float16* L_vt   = (_Float16*)alloc(128 * 256 * 2);
  _Float16* msg_s = msg;
  _Float16* msg_v = msg + (size_t)NNODE * 256;

  const float invsqrtc = 0.08838834764831845f;   // 1/sqrt(128)
  const float outscale = 0.0625f / 20.0f;        // 1/sqrt(256) / AVG_NEIGH

  auto tr = [&](const float* s, _Float16* d, int K, int Nn, float sc) {
    int n = K * Nn;
    k_transpose_f16<<<(n + 255) / 256, 256, 0, stream>>>(s, d, K, Nn, sc);
  };
  tr(Ws,    Wst,   128, 128, invsqrtc);
  tr(Wup_s, WupSt, 128, 128, invsqrtc);
  tr(Wup_v, WupVt, 128, 128, invsqrtc);
  tr(W1,    W1t,   256, 128, 1.0f);   // first 256 rows; rows 256/257 applied in epilogue
  tr(W2,    W2t,   128, 128, 1.0f);
  tr(W3,    W3t,   128, 512, 1.0f);
  tr(L_s,   L_st,  256, 128, outscale);
  tr(L_v,   L_vt,  256, 128, outscale);

  // zero f16 message accumulators (4*N*256 halves == 5.12M dwords)
  k_zero<<<4096, 256, 0, stream>>>((float*)msg, 4 * NNODE * 256 / 2);

  k_node_up<<<(NNODE + 127) / 128, 256, 0, stream>>>(
      node_feats, Wst, WupSt, WupVt, ns_f16, s_up, v_up);

  const int* sender   = edge_index;
  const int* receiver = edge_index + NEDGE;
  k_edge<<<(NEDGE / 16 + 7) / 8, 256, 0, stream>>>(
      ns_f16, s_up, v_up, edge_attrs, edge_feats, lengths,
      sender, receiver, W1, b1, b2, W1t, W2t, W3t, msg_s, msg_v);

  k_out<<<(NNODE + 127) / 128, 256, 0, stream>>>(msg_s, msg_v, L_st, L_vt, out);
}
